// MicroringBasisBase_78159814853130
// MI455X (gfx1250) — matrix operand order repaired and checked by reading, not compiled, not run on hardware
//
#include <hip/hip_runtime.h>
#include <math.h>

typedef _Float16 v16h __attribute__((ext_vector_type(16)));
typedef _Float16 h8   __attribute__((ext_vector_type(8)));
typedef float    v8f  __attribute__((ext_vector_type(8)));

#define B_DIM   4096
#define IN_DIM  512
#define OUT_DIM 512
#define NRING   16
#define KBASIS  (IN_DIM * NRING)          // 8192
#define KAUG    (KBASIS + IN_DIM)         // 8704 (basis ++ residual)
#define BM      64                        // M tile
#define BN      128                       // N tile
#define KC      256                       // K chunk
#define KPAD    8
#define LDB     (KC + KPAD)               // 264 halves per LDS row

// sched_group_barrier masks
#define SGB_MFMA_WMMA 0x008
#define SGB_DS_READ   0x100

union Frag { v16h v; h8 h[2]; };

// =====================================================================
// One-time pre-conversion: Wt[n][k] (f16, k-major rows of 8704)
//   k <  8192 : Wt[n][i*16+r] = coeffs[i][n][r]
//   k >= 8192 : Wt[n][8192+kk] = base_weight[kk][n]
// =====================================================================
__global__ __launch_bounds__(256)
void preconvert_kernel(const float* __restrict__ coeffs,
                       const float* __restrict__ bw,
                       _Float16* __restrict__ wt)
{
    const int gid = blockIdx.x * blockDim.x + threadIdx.x;
    const int NA  = OUT_DIM * (KBASIS / 8);            // 524288 groups of 8
    if (gid < NA) {
        const int n  = gid >> 10;
        const int k  = (gid & 1023) * 8;
        const int i  = k >> 4;
        const int r  = k & 15;                          // 0 or 8
        const float4* src = (const float4*)(coeffs + ((size_t)i * OUT_DIM + n) * NRING + r);
        float4 f0 = src[0], f1 = src[1];
        h8 v;
        v[0] = (_Float16)f0.x; v[1] = (_Float16)f0.y;
        v[2] = (_Float16)f0.z; v[3] = (_Float16)f0.w;
        v[4] = (_Float16)f1.x; v[5] = (_Float16)f1.y;
        v[6] = (_Float16)f1.z; v[7] = (_Float16)f1.w;
        *(h8*)(wt + (size_t)n * KAUG + k) = v;
    } else {
        const int g = gid - NA;                         // 32768 groups of 8 (along n)
        if (g < IN_DIM * (OUT_DIM / 8)) {
            const int kk = g >> 6;
            const int n8 = (g & 63) * 8;
            const float4* src = (const float4*)(bw + (size_t)kk * OUT_DIM + n8);
            float4 f0 = src[0], f1 = src[1];
            float vals[8] = {f0.x, f0.y, f0.z, f0.w, f1.x, f1.y, f1.z, f1.w};
            #pragma unroll
            for (int j = 0; j < 8; ++j)
                wt[(size_t)(n8 + j) * KAUG + KBASIS + kk] = (_Float16)vals[j];
        }
    }
}

// =====================================================================
// Fused basis + GEMM kernel.  PRE=true: B-slabs copied from pre-converted
// f16 Wt.  PRE=false: B-slabs converted from f32 inputs per chunk.
// =====================================================================
template <bool PRE>
__global__ __launch_bounds__(256)
void mrr_kernel(const float* __restrict__ x,
                const float* __restrict__ coeffs,
                const float* __restrict__ bw,
                const _Float16* __restrict__ wt,
                float* __restrict__ out)
{
    // ---- physics constants (double, evaluated once per thread) ----
    const double PI   = 3.14159265358979323846;
    const double Ld   = 2.0 * PI * 30.0e-6;
    const double WL0d = 1550.0e-9;
    const double Ad   = exp(-(3.0 * (Ld * 100.0) / 20.0) * 2.302585092994045684);
    const double Rtd  = sqrt(1.0 - 0.2);
    const double T0d  = 3209.0;                        // t = K0/wl lies in [3201, 3218]
    const float  K0F  = (float)(2.0 * PI * Ld * 4.2);  // 2*pi*L*NG
    const float  T0F  = (float)T0d;
    const float  CN   = (float)(Ad * Ad + Rtd * Rtd);
    const float  CD   = (float)(1.0 + (Rtd * Ad) * (Rtd * Ad));
    const float  A1   = (float)(2.0 * Rtd * Ad);
    const float  DK   = CD - CN;
    const double P0   = -2.0 * PI * Ld * (4.2 - 2.34) / WL0d;

    float cp[NRING], sp[NRING];
    #pragma unroll
    for (int r = 0; r < NRING; ++r) {
        double pr = P0 + T0d + (-PI + (double)r * (2.0 * PI / 15.0));
        double s, c; sincos(pr, &s, &c);
        cp[r] = (float)c; sp[r] = (float)s;
    }

    __shared__ _Float16 As[BM][LDB];
    __shared__ _Float16 Bs[BN][LDB];

    const int tid  = threadIdx.x;
    const int lane = tid & 31;
    const int wave = tid >> 5;
    const int wm   = wave & 3;              // 4 waves along M (16 rows each)
    const int wn   = wave >> 2;             // 2 waves along N (64 cols each)
    const int bm   = blockIdx.y * BM;
    const int n0   = blockIdx.x * BN;

    v8f acc[4];
    #pragma unroll
    for (int t = 0; t < 4; ++t)
        #pragma unroll
        for (int j = 0; j < 8; ++j) acc[t][j] = 0.0f;

    // ---- fragment loaders (ISA VGPR layouts) ----
    auto load_a = [&](Frag& f, int ks) {
        const int row = wm * 16 + (lane & 15);
        const int kh  = (lane >> 4) * 8;
        f.h[0] = *(const h8*)&As[row][ks * 32 + kh];
        f.h[1] = *(const h8*)&As[row][ks * 32 + 16 + kh];
    };
    auto load_b = [&](Frag& f, int s) {
        const int col = wn * 64 + (s & 3) * 16 + (lane & 15);
        const int kb  = (s >> 2) * 32 + (lane >> 4) * 8;
        f.h[0] = *(const h8*)&Bs[col][kb];
        f.h[1] = *(const h8*)&Bs[col][kb + 16];
    };

    // ---- software-pipelined MMA over one resident K-chunk ----
    // A double-buffered, B ring-of-4 (prefetch distance 3).  IGLP group
    // barriers pin the [DS-reads][WMMA] interleave so the scheduler cannot
    // sink the ring loads back to their consumers (which produced
    // s_wait_dscnt 0x0 per WMMA in previous builds).
    auto mma_chunk = [&]() {
        Frag a[2], b[4];
        load_a(a[0], 0);
        load_b(b[0], 0); load_b(b[1], 1); load_b(b[2], 2);
        #pragma unroll
        for (int s = 0; s < 32; ++s) {          // s = ks*4 + t
            const int ks = s >> 2, t = s & 3;
            if (t == 0 && ks < 7) load_a(a[(ks + 1) & 1], ks + 1);
            if (s + 3 < 32)       load_b(b[(s + 3) & 3], s + 3);
            acc[t] = __builtin_amdgcn_wmma_f32_16x16x32_f16(
                false, a[ks & 1].v, false, b[s & 3].v, (short)0, acc[t], false, false);
            __builtin_amdgcn_sched_group_barrier(SGB_DS_READ,   4, 0);
            __builtin_amdgcn_sched_group_barrier(SGB_MFMA_WMMA, 1, 0);
        }
    };

    auto cvt16 = [](const float4* __restrict__ src, h8& lo, h8& hi) {
        float4 f0 = src[0], f1 = src[1], f2 = src[2], f3 = src[3];
        lo[0]=(_Float16)f0.x; lo[1]=(_Float16)f0.y; lo[2]=(_Float16)f0.z; lo[3]=(_Float16)f0.w;
        lo[4]=(_Float16)f1.x; lo[5]=(_Float16)f1.y; lo[6]=(_Float16)f1.z; lo[7]=(_Float16)f1.w;
        hi[0]=(_Float16)f2.x; hi[1]=(_Float16)f2.y; hi[2]=(_Float16)f2.z; hi[3]=(_Float16)f2.w;
        hi[4]=(_Float16)f3.x; hi[5]=(_Float16)f3.y; hi[6]=(_Float16)f3.z; hi[7]=(_Float16)f3.w;
    };

    // ================= unified augmented-K loop: 32 basis + 2 residual chunks ========
    for (int ch = 0; ch < KAUG / KC; ++ch) {
        const int kbase = ch * KC;

        // ---------- A-slab ----------
        if (ch < KBASIS / KC) {
            // basis(x[bm+row, i0+il], ring r) -> As[row][il*16 + r]
            const int i0 = ch * 16;
            for (int p = tid; p < BM * 16; p += 256) {       // 4 iters
                const int row = p >> 4, il = p & 15;
                float xv = x[(size_t)(bm + row) * IN_DIM + i0 + il];
                float xc = fminf(fmaxf(xv, -1.0f), 1.0f);
                float wl = 1546.0e-9f + (xc + 1.0f) * 4.0e-9f;
                // u = K0/wl - T0, cancellation-free via FMA
                float u  = fmaf(-T0F, wl, K0F) * __builtin_amdgcn_rcpf(wl);
                float st, ct;
                __sincosf(u, &st, &ct);
                h8 lo, hi;
                #pragma unroll
                for (int r = 0; r < 8; ++r) {
                    float c   = fmaf(ct, cp[r], -st * sp[r]);
                    float den = fmaf(-A1, c, CD);
                    lo[r] = (_Float16)fmaf(-DK, __builtin_amdgcn_rcpf(den), 1.0f);
                }
                #pragma unroll
                for (int r = 0; r < 8; ++r) {
                    float c   = fmaf(ct, cp[r + 8], -st * sp[r + 8]);
                    float den = fmaf(-A1, c, CD);
                    hi[r] = (_Float16)fmaf(-DK, __builtin_amdgcn_rcpf(den), 1.0f);
                }
                *(h8*)&As[row][il * 16]     = lo;
                *(h8*)&As[row][il * 16 + 8] = hi;
            }
        } else {
            // residual: As[row][k] = (f16) x[bm+row, c0+k]
            const int c0 = (ch - KBASIS / KC) * KC;
            for (int p = tid; p < BM * 16; p += 256) {
                const int row = p >> 4, g = p & 15;
                const float4* src = (const float4*)(x + (size_t)(bm + row) * IN_DIM + c0 + g * 16);
                h8 lo, hi; cvt16(src, lo, hi);
                *(h8*)&As[row][g * 16]     = lo;
                *(h8*)&As[row][g * 16 + 8] = hi;
            }
        }

        // ---------- B-slab ----------
        if constexpr (PRE) {
            // plain f16 copy from pre-converted augmented weights (L2-resident)
            for (int p = tid; p < BN * 16; p += 256) {       // 8 iters
                const int n = p >> 4, kg = (p & 15) * 16;
                const _Float16* src = wt + (size_t)(n0 + n) * KAUG + kbase + kg;
                *(h8*)&Bs[n][kg]     = *(const h8*)src;
                *(h8*)&Bs[n][kg + 8] = *(const h8*)(src + 8);
                // warm WGP$/L2 for next chunk's slab (global_prefetch_b8)
                if (kbase + KC < KAUG)
                    __builtin_prefetch((const void*)(src + KC), 0, 1);
            }
        } else {
            if (ch < KBASIS / KC) {
                const int i0 = ch * 16;
                for (int p = tid; p < BN * 16; p += 256) {   // 8 iters
                    const int n = p & 127, il = p >> 7;
                    const float4* src =
                        (const float4*)(coeffs + ((size_t)(i0 + il) * OUT_DIM + (n0 + n)) * NRING);
                    h8 lo, hi; cvt16(src, lo, hi);
                    *(h8*)&Bs[n][il * 16]     = lo;
                    *(h8*)&Bs[n][il * 16 + 8] = hi;
                }
            } else {
                const int c0 = (ch - KBASIS / KC) * KC;
                for (int p = tid; p < KC * 32; p += 256) {   // 32 iters
                    const int k = p >> 5, nf = p & 31;
                    float4 w = *(const float4*)(bw + (size_t)(c0 + k) * OUT_DIM + n0 + nf * 4);
                    Bs[nf * 4 + 0][k] = (_Float16)w.x;
                    Bs[nf * 4 + 1][k] = (_Float16)w.y;
                    Bs[nf * 4 + 2][k] = (_Float16)w.z;
                    Bs[nf * 4 + 3][k] = (_Float16)w.w;
                }
            }
        }

        __syncthreads();
        mma_chunk();
        __syncthreads();
    }

    // ================= epilogue: f32 C/D layout -> global =================
    #pragma unroll
    for (int t = 0; t < 4; ++t) {
        const int col = n0 + wn * 64 + t * 16 + (lane & 15);
        const int rb  = bm + wm * 16 + ((lane >> 4) << 3);
        #pragma unroll
        for (int v = 0; v < 8; ++v)
            out[(size_t)(rb + v) * OUT_DIM + col] = acc[t][v];
    }

    if (blockIdx.x == 0 && blockIdx.y == 0 && tid == 0)
        out[(size_t)B_DIM * OUT_DIM] = 0.0f;   // kl = 0
}

extern "C" void kernel_launch(void* const* d_in, const int* in_sizes, int n_in,
                              void* d_out, int out_size, void* d_ws, size_t ws_size,
                              hipStream_t stream) {
    (void)in_sizes; (void)n_in; (void)out_size;
    const float* x      = (const float*)d_in[0];   // [4096, 512]
    const float* coeffs = (const float*)d_in[1];   // [512, 512, 16]
    const float* bw     = (const float*)d_in[2];   // [512, 512]
    float* out = (float*)d_out;                    // [4096*512 + 1]

    const size_t WT_BYTES = (size_t)OUT_DIM * KAUG * sizeof(_Float16);  // ~8.9 MB
    dim3 grid(OUT_DIM / BN, B_DIM / BM);           // (4, 64) = 256 workgroups

    if (ws_size >= WT_BYTES) {
        _Float16* wt = (_Float16*)d_ws;
        const int groups = OUT_DIM * (KBASIS / 8) + IN_DIM * (OUT_DIM / 8); // 557056
        preconvert_kernel<<<(groups + 255) / 256, 256, 0, stream>>>(coeffs, bw, wt);
        mrr_kernel<true><<<grid, 256, 0, stream>>>(x, coeffs, bw, wt, out);
    } else {
        mrr_kernel<false><<<grid, 256, 0, stream>>>(x, coeffs, bw, nullptr, out);
    }
}
